// E63cHDependentGatesCell_4887672783120
// MI455X (gfx1250) — compile-verified, matrix operand order checked by reading, not run on hardware
//
#include <hip/hip_runtime.h>

// ---------------------------------------------------------------------------
// E63c h-dependent-gates RNN cell for MI455X (gfx1250, wave32, WMMA).
//
//  T=1024, B=32, D=1024.
//  Phase 1: convert x + 6 weight matrices fp32 -> bf16 (workspace).
//  Phase 2: projection GEMM proj[z] = x @ W_z^T (z = alpha,beta,x) via
//           v_wmma_f32_16x16x32_bf16, 64x64 block tiles, 4 waves/block.
//  Phase 3: persistent recurrent kernel, 32 WGs x 128 threads.
//           - U_alpha/U_beta/W_h column slices LDS-resident (~194 KB/WGP).
//           - h ping-pong kept in global in WMMA A-fragment layout, so the
//             A operand loads straight from L2 with 2 coalesced b128 loads.
//           - per wave: ONE fused k-loop, 3 independent accumulators
//             (alpha/beta/v) -> WMMA hazard slots filled, 3x less A traffic.
//           - global_prefetch of the step's xproj / h[t] lines up front.
//           - grid barrier per step (release fence + atomic + acquire fence).
// ---------------------------------------------------------------------------

#define T_ 1024
#define B_ 32
#define D_ 1024

#define NWG     32      // workgroups in persistent kernel
#define SLICE   32      // e-columns per WG
#define RBLOCK  128     // 4 waves (wave32)

#define ROWB 2064                 // padded LDS row bytes for 1024 bf16 (+16B)
#define MATB (32 * ROWB)          // one 32-row weight slice: 66048 B
#define SP   33                   // f32 staging pitch

typedef float  v8f   __attribute__((ext_vector_type(8)));
typedef __bf16 v16bf __attribute__((ext_vector_type(16)));

union V32B { uint4 q[2]; v16bf v; };

// load 16 contiguous bf16 (32 B, 16B-aligned) from LDS/generic
__device__ __forceinline__ v16bf ld32_lds(const char* p) {
    V32B u;
    u.q[0] = *(const uint4*)(p);
    u.q[1] = *(const uint4*)(p + 16);
    return u.v;
}
// load 16 contiguous bf16 from global
__device__ __forceinline__ v16bf ld32_g(const __bf16* p) {
    V32B u;
    u.q[0] = *(const uint4*)(p);
    u.q[1] = *(const uint4*)(p + 8);
    return u.v;
}

__device__ __forceinline__ float sigmoid_f(float x) {
    return 1.0f / (1.0f + __expf(-x));
}

// ---------------------------------------------------------------------------
// fp32 -> bf16 convert
// ---------------------------------------------------------------------------
__global__ void cvt_f32_bf16(const float* __restrict__ s,
                             __bf16* __restrict__ d, int n) {
    int i = blockIdx.x * blockDim.x + threadIdx.x;
    if (i < n) d[i] = (__bf16)s[i];
}

// ---------------------------------------------------------------------------
// init: h[0] = 0, hbf ping = 0, barrier counter = 0  (re-run every launch)
// ---------------------------------------------------------------------------
__global__ void init_state(float* __restrict__ h0,
                           __bf16* __restrict__ hbf0,
                           unsigned* __restrict__ ctr) {
    int i = blockIdx.x * blockDim.x + threadIdx.x;
    if (i < B_ * D_) {
        h0[i]   = 0.0f;
        hbf0[i] = (__bf16)0.0f;
    }
    if (i == 0) *ctr = 0u;
}

// ---------------------------------------------------------------------------
// Projection GEMM: proj[z][m, e] = sum_d xb[m, d] * W_z[e, d]   (bf16 out)
// grid = (M/64, D/64, 3), block = 128.  Each wave -> 32x32 tile (4 WMMAs/kt).
// ---------------------------------------------------------------------------
extern "C" __global__ void __launch_bounds__(128)
proj_gemm(const __bf16* __restrict__ xb,    // [M, D]   M = T*B
          const __bf16* __restrict__ wts,   // [3][D, D]  Wa, Wb, Wx
          __bf16* __restrict__ proj)        // [3][M, D]
{
    const int M = T_ * B_;
    const int z = blockIdx.z;
    const __bf16* W = wts  + (size_t)z * D_ * D_;
    __bf16*       C = proj + (size_t)z * M  * D_;

    const int tid   = threadIdx.x;
    const int lane  = tid & 31;
    const int wv    = tid >> 5;
    const int lrow  = lane & 15;
    const int lhalf = lane >> 4;
    const int m0 = blockIdx.x * 64 + (wv >> 1) * 32;
    const int n0 = blockIdx.y * 64 + (wv & 1) * 32;

    v8f acc[2][2] = {};
    for (int kt = 0; kt < D_ / 32; ++kt) {
        const int k = kt * 32 + lhalf * 16;
        v16bf a0 = ld32_g(xb + (size_t)(m0 + lrow)      * D_ + k);
        v16bf a1 = ld32_g(xb + (size_t)(m0 + 16 + lrow) * D_ + k);
        v16bf b0 = ld32_g(W  + (size_t)(n0 + lrow)      * D_ + k);
        v16bf b1 = ld32_g(W  + (size_t)(n0 + 16 + lrow) * D_ + k);
        acc[0][0] = __builtin_amdgcn_wmma_f32_16x16x32_bf16(false, a0, false, b0, (short)0, acc[0][0], false, false);
        acc[0][1] = __builtin_amdgcn_wmma_f32_16x16x32_bf16(false, a0, false, b1, (short)0, acc[0][1], false, false);
        acc[1][0] = __builtin_amdgcn_wmma_f32_16x16x32_bf16(false, a1, false, b0, (short)0, acc[1][0], false, false);
        acc[1][1] = __builtin_amdgcn_wmma_f32_16x16x32_bf16(false, a1, false, b1, (short)0, acc[1][1], false, false);
    }
#pragma unroll
    for (int mt = 0; mt < 2; ++mt)
#pragma unroll
        for (int nt = 0; nt < 2; ++nt) {
            const int row = m0 + mt * 16 + lhalf * 8;
            const int col = n0 + nt * 16 + lrow;
#pragma unroll
            for (int r = 0; r < 8; ++r)
                C[(size_t)(row + r) * D_ + col] = (__bf16)acc[mt][nt][r];
        }
}

// ---------------------------------------------------------------------------
// Persistent recurrent kernel.  grid = NWG, block = RBLOCK (4 waves).
// WG g owns e-slice [g*32, g*32+32).
//
// h ping-pong (global, bf16) is stored in WMMA A-fragment layout:
//   element (b, e):  mt = b>>4, r = b&15, kt = e>>5, half = (e>>4)&1,
//                    elem = e&15, lane = half*16 + r
//   index = ((mt*32 + kt)*32 + lane)*16 + elem
// so a wave's A tile for (mt, kt) is 32 lanes x 16 contiguous bf16.
//
// LDS (dynamic, ~206 KB): 3 weight slices [3][32][ROWB] + staging [3][32][SP].
// ---------------------------------------------------------------------------
extern "C" __global__ void __launch_bounds__(RBLOCK)
rnn_persistent(const __bf16* __restrict__ xproj,  // [3][T*B*D]  ax, bx, wx
               const __bf16* __restrict__ wrec,   // [3][D*D]    Ua, Ub, Wh
               const float* __restrict__ b_alpha,
               const float* __restrict__ b_beta,
               const float* __restrict__ b_out,
               float* __restrict__ outbuf,        // [T*B*D] outputs, then [(T+1)*B*D] h
               __bf16* __restrict__ hbf,          // [2][B*D] ping-pong, A-fragment layout
               unsigned* __restrict__ ctr)
{
    extern __shared__ char smem[];
    char*  lds_wB = smem;                       // 3 * MATB
    float* stage  = (float*)(smem + 3 * MATB);  // [3][32][SP]

    const int g     = blockIdx.x;
    const int e0    = g * SLICE;
    const int tid   = threadIdx.x;
    const int lane  = tid & 31;
    const int wv    = tid >> 5;
    const int lrow  = lane & 15;
    const int lhalf = lane >> 4;
    const size_t SLAB = (size_t)T_ * B_ * D_;

    float* h_arr = outbuf + SLAB;   // h region: (T+1) x B x D

    // ---- one-time: weight slices -> LDS (16 B moves) ----
    for (int i = tid; i < 3 * 32 * (D_ / 8); i += RBLOCK) {
        const int m  = i / (32 * (D_ / 8));
        const int r  = (i / (D_ / 8)) & 31;
        const int c8 = i % (D_ / 8);
        const uint4 v = *(const uint4*)(wrec + (size_t)m * D_ * D_ +
                                        (size_t)(e0 + r) * D_ + c8 * 8);
        *(uint4*)(lds_wB + m * MATB + r * ROWB + c8 * 16) = v;
    }

    // ---- per-thread invariants for the elementwise phase ----
    const int el   = tid & 31;         // e column within slice (fixed per thread)
    const int e    = e0 + el;
    const int b0i  = tid >> 5;         // first batch row, step 4
    const float ba_r = b_alpha[e];
    const float bb_r = b_beta [e];
    const float bo_r = b_out  [e];
    const int kt_w   = e >> 5;         // h-fragment coords for column e
    const int half_w = (e >> 4) & 1;
    const int elem_w = e & 15;

    // ---- per-wave GEMM tile invariants ----
    const int mt = (wv >> 1) & 1;      // batch-tile of this wave
    const int nt = wv & 1;             // n-tile of this wave
    const char* wB0 = lds_wB + 0 * MATB + (nt * 16 + lrow) * ROWB + lhalf * 32;
    const char* wB1 = lds_wB + 1 * MATB + (nt * 16 + lrow) * ROWB + lhalf * 32;
    const char* wB2 = lds_wB + 2 * MATB + (nt * 16 + lrow) * ROWB + lhalf * 32;

    __syncthreads();

    for (int t = 0; t < T_; ++t) {
        const size_t tslab = (size_t)t * (B_ * D_);

        // ---- prefetch this step's streaming operands (global_prefetch_b8) ----
        if (tid < 96) {
            const int arr = tid >> 5, b = tid & 31;
            __builtin_prefetch(xproj + (size_t)arr * SLAB + tslab + (size_t)b * D_ + e0, 0, 1);
        }
        if (tid < 32) {
            __builtin_prefetch(h_arr + tslab + (size_t)tid * D_ + e0, 0, 1);
            __builtin_prefetch(h_arr + tslab + (size_t)tid * D_ + e0 + 16, 0, 1);
        }

        // ---- fused GEMM: 3 matrices, 3 independent accumulators per wave ----
        const __bf16* aBase = hbf + (size_t)(t & 1) * (B_ * D_)
                                  + (size_t)mt * (32 * 32 * 16) + (size_t)lane * 16;
        v8f acc0 = {}, acc1 = {}, acc2 = {};
        for (int kt = 0; kt < D_ / 32; ++kt) {
            v16bf a  = ld32_g(aBase + (size_t)kt * 512);    // A fragment (shared)
            v16bf w0 = ld32_lds(wB0 + kt * 64);
            v16bf w1 = ld32_lds(wB1 + kt * 64);
            v16bf w2 = ld32_lds(wB2 + kt * 64);
            acc0 = __builtin_amdgcn_wmma_f32_16x16x32_bf16(false, a, false, w0, (short)0, acc0, false, false);
            acc1 = __builtin_amdgcn_wmma_f32_16x16x32_bf16(false, a, false, w1, (short)0, acc1, false, false);
            acc2 = __builtin_amdgcn_wmma_f32_16x16x32_bf16(false, a, false, w2, (short)0, acc2, false, false);
        }
        {
            const int brow = mt * 16 + lhalf * 8;
            const int col  = nt * 16 + lrow;
#pragma unroll
            for (int r = 0; r < 8; ++r) stage[0 * (32 * SP) + (brow + r) * SP + col] = acc0[r];
#pragma unroll
            for (int r = 0; r < 8; ++r) stage[1 * (32 * SP) + (brow + r) * SP + col] = acc1[r];
#pragma unroll
            for (int r = 0; r < 8; ++r) stage[2 * (32 * SP) + (brow + r) * SP + col] = acc2[r];
        }
        __syncthreads();

        // ---- fused gating elementwise over this WG's 32x32 slice ----
        __bf16* hdst = hbf + (size_t)((t + 1) & 1) * (B_ * D_);
#pragma unroll
        for (int j = 0; j < 8; ++j) {
            const int b = b0i + j * 4;
            const size_t gidx = tslab + (size_t)b * D_ + e;

            const float hp   = h_arr[gidx];                        // h[t]
            const float apre = stage[0 * (32 * SP) + b * SP + el]
                             + (float)xproj[0 * SLAB + gidx] + ba_r;
            const float bpre = stage[1 * (32 * SP) + b * SP + el]
                             + (float)xproj[1 * SLAB + gidx] + bb_r;
            const float vpre = stage[2 * (32 * SP) + b * SP + el]
                             + (float)xproj[2 * SLAB + gidx] + bo_r;

            const float alpha = sigmoid_f(apre);
            const float beta  = sigmoid_f(bpre);
            const float v     = tanhf(vpre);
            const float hn    = alpha * hp + beta * v;

            h_arr[gidx + (size_t)B_ * D_] = hn;                    // h[t+1]
            outbuf[gidx] = hn * hn * sigmoid_f(hn);                // h * silu(h)

            const int lane_w = half_w * 16 + (b & 15);
            const int mt_w   = b >> 4;
            hdst[((mt_w * 32 + kt_w) * 32 + lane_w) * 16 + elem_w] = (__bf16)hn;
        }

        // ---- grid-wide barrier: release h writes, acquire others' h ----
        __threadfence();
        __syncthreads();
        if (tid == 0) {
            __hip_atomic_fetch_add(ctr, 1u, __ATOMIC_ACQ_REL, __HIP_MEMORY_SCOPE_AGENT);
            const unsigned tgt = (unsigned)(t + 1) * NWG;
            while (__hip_atomic_load(ctr, __ATOMIC_ACQUIRE, __HIP_MEMORY_SCOPE_AGENT) < tgt) {
                __builtin_amdgcn_s_sleep(2);
            }
        }
        __syncthreads();
        __threadfence();   // acquire side: invalidate stale near-cache lines
    }
}

// ---------------------------------------------------------------------------
// host launch
// ---------------------------------------------------------------------------
extern "C" void kernel_launch(void* const* d_in, const int* in_sizes, int n_in,
                              void* d_out, int out_size, void* d_ws, size_t ws_size,
                              hipStream_t stream) {
    (void)in_sizes; (void)n_in; (void)out_size; (void)ws_size;

    const float* x   = (const float*)d_in[0];
    const float* Wa  = (const float*)d_in[1];
    const float* Ua  = (const float*)d_in[2];
    const float* ba  = (const float*)d_in[3];
    const float* Wb  = (const float*)d_in[4];
    const float* Ub  = (const float*)d_in[5];
    const float* bb  = (const float*)d_in[6];
    const float* Wh  = (const float*)d_in[7];
    const float* Wx  = (const float*)d_in[8];
    const float* bo  = (const float*)d_in[9];

    const size_t NX = (size_t)T_ * B_ * D_;     // 33,554,432
    const size_t DD = (size_t)D_ * D_;

    // workspace layout (bf16 units): x_bf | proj[3] | Wpre[3] | Wrec[3] | hbf[2] | ctr
    __bf16* ws    = (__bf16*)d_ws;
    __bf16* x_bf  = ws;
    __bf16* proj  = ws + NX;
    __bf16* wpre  = ws + 4 * NX;            // Wa, Wb, Wx
    __bf16* wrec  = wpre + 3 * DD;          // Ua, Ub, Wh
    __bf16* hbf   = wrec + 3 * DD;          // 2 * B * D (A-fragment layout)
    unsigned* ctr = (unsigned*)(hbf + 2 * (size_t)B_ * D_);

    const int thr = 256;
    cvt_f32_bf16<<<(unsigned)((NX + thr - 1) / thr), thr, 0, stream>>>(x,  x_bf,        (int)NX);
    cvt_f32_bf16<<<(unsigned)((DD + thr - 1) / thr), thr, 0, stream>>>(Wa, wpre + 0*DD, (int)DD);
    cvt_f32_bf16<<<(unsigned)((DD + thr - 1) / thr), thr, 0, stream>>>(Wb, wpre + 1*DD, (int)DD);
    cvt_f32_bf16<<<(unsigned)((DD + thr - 1) / thr), thr, 0, stream>>>(Wx, wpre + 2*DD, (int)DD);
    cvt_f32_bf16<<<(unsigned)((DD + thr - 1) / thr), thr, 0, stream>>>(Ua, wrec + 0*DD, (int)DD);
    cvt_f32_bf16<<<(unsigned)((DD + thr - 1) / thr), thr, 0, stream>>>(Ub, wrec + 1*DD, (int)DD);
    cvt_f32_bf16<<<(unsigned)((DD + thr - 1) / thr), thr, 0, stream>>>(Wh, wrec + 2*DD, (int)DD);

    // proj[z] = x @ Wz^T ; grid (M/64, D/64, 3)
    proj_gemm<<<dim3((T_ * B_) / 64, D_ / 64, 3), 128, 0, stream>>>(x_bf, wpre, proj);

    float* out = (float*)d_out;
    init_state<<<(B_ * D_ + 255) / 256, 256, 0, stream>>>(out + NX, hbf, ctr);

    const size_t ldsB = 3 * (size_t)MATB + 3 * 32 * SP * 4;   // 198144 + 12672 B
    rnn_persistent<<<NWG, RBLOCK, ldsB, stream>>>(proj, wrec, ba, bb, bo, out, hbf, ctr);
}
